// STPv3Standalone_3126736192071
// MI455X (gfx1250) — matrix operand order repaired and checked by reading, not compiled, not run on hardware
//
#include <hip/hip_runtime.h>
#include <hip/hip_bf16.h>

// ---------------------------------------------------------------------------
// Problem constants (from reference setup_inputs)
// ---------------------------------------------------------------------------
#define Bv  4
#define Tv  2048
#define Dv  1024
#define Hv  16
#define DKv 64
#define Vv  8192
#define Lv  2
#define BT  (Bv * Tv)          // 8192 rows
#define EPS 1e-5f

typedef __attribute__((ext_vector_type(16))) __bf16 v16bf;
typedef __attribute__((ext_vector_type(8)))  float  v8f;
typedef __attribute__((ext_vector_type(4)))  unsigned int v4u;
typedef __attribute__((ext_vector_type(8)))  int v8i;
typedef __attribute__((ext_vector_type(4)))  int v4i;

#if defined(__has_builtin)
#  if __has_builtin(__builtin_amdgcn_tensor_load_to_lds) && \
      __has_builtin(__builtin_amdgcn_s_wait_tensorcnt)
#    define USE_TDM 1
#  endif
#endif
#ifndef USE_TDM
#  define USE_TDM 0
#endif

union FragAB {
    uint4 q[2];
    v16bf v;
};

__device__ __forceinline__ unsigned short f2b(float f) {
    unsigned int u = __float_as_uint(f);
    unsigned int r = u + 0x7FFFu + ((u >> 16) & 1u);   // round-to-nearest-even
    return (unsigned short)(r >> 16);
}

// ---------------------------------------------------------------------------
// Embedding gather: X[bt,:] = E[ids[bt],:]
// ---------------------------------------------------------------------------
__global__ __launch_bounds__(256) void embed_kernel(const int* __restrict__ ids,
                                                    const float* __restrict__ E,
                                                    float* __restrict__ X) {
    int bt = blockIdx.x;
    int id = ids[bt];
    const float4* src = (const float4*)(E + (size_t)id * Dv);
    float4*       dst = (float4*)(X + (size_t)bt * Dv);
    dst[threadIdx.x] = src[threadIdx.x];
}

// ---------------------------------------------------------------------------
// LayerNorm over D=1024; optional bf16 output for GEMM feeding
// ---------------------------------------------------------------------------
__global__ __launch_bounds__(256) void ln1024_kernel(const float* __restrict__ in,
                                                     const float* __restrict__ g,
                                                     const float* __restrict__ b,
                                                     float* __restrict__ out32,
                                                     unsigned short* __restrict__ out16) {
    int row = blockIdx.x;
    int tid = threadIdx.x;
    float4 x = ((const float4*)(in + (size_t)row * Dv))[tid];
    float s  = x.x + x.y + x.z + x.w;
    float s2 = x.x * x.x + x.y * x.y + x.z * x.z + x.w * x.w;
    #pragma unroll
    for (int off = 1; off < 32; off <<= 1) {
        s  += __shfl_xor(s,  off, 32);
        s2 += __shfl_xor(s2, off, 32);
    }
    __shared__ float ss[8], ss2[8];
    if ((tid & 31) == 0) { ss[tid >> 5] = s; ss2[tid >> 5] = s2; }
    __syncthreads();
    float S = 0.f, S2 = 0.f;
    #pragma unroll
    for (int r = 0; r < 8; ++r) { S += ss[r]; S2 += ss2[r]; }
    float mean = S * (1.f / Dv);
    float var  = S2 * (1.f / Dv) - mean * mean;
    float rstd = rsqrtf(var + EPS);
    float4 gv = ((const float4*)g)[tid];
    float4 bv = ((const float4*)b)[tid];
    float4 o;
    o.x = (x.x - mean) * rstd * gv.x + bv.x;
    o.y = (x.y - mean) * rstd * gv.y + bv.y;
    o.z = (x.z - mean) * rstd * gv.z + bv.z;
    o.w = (x.w - mean) * rstd * gv.w + bv.w;
    if (out32) ((float4*)(out32 + (size_t)row * Dv))[tid] = o;
    if (out16) {
        ushort4 o16;
        o16.x = f2b(o.x); o16.y = f2b(o.y); o16.z = f2b(o.z); o16.w = f2b(o.w);
        ((ushort4*)(out16 + (size_t)row * Dv))[tid] = o16;
    }
}

// ---------------------------------------------------------------------------
// Causal depthwise conv (K=3, left pad 2) fused with residual h = hln + conv,
// emitting bf16 GEMM operand directly.
// ---------------------------------------------------------------------------
__global__ __launch_bounds__(256) void conv_kernel(const float* __restrict__ hln,
                                                   const float* __restrict__ w,   // (3, D)
                                                   const float* __restrict__ cb,  // (D)
                                                   unsigned short* __restrict__ out) {
    size_t idx = (size_t)blockIdx.x * 256 + threadIdx.x;   // over B*T*D
    int d = (int)(idx & (Dv - 1));
    size_t bt = idx >> 10;
    int t = (int)(bt & (Tv - 1));
    float h0  = hln[idx];
    float hm1 = (t >= 1) ? hln[idx - Dv] : 0.f;
    float hm2 = (t >= 2) ? hln[idx - 2 * Dv] : 0.f;
    float hc = w[d] * hm2 + w[Dv + d] * hm1 + w[2 * Dv + d] * h0 + cb[d];
    out[idx] = f2b(h0 + hc);
}

// ---------------------------------------------------------------------------
// f32 -> bf16 bulk convert (4 elements / thread)
// ---------------------------------------------------------------------------
__global__ __launch_bounds__(256) void cvt_f2b_kernel(const float* __restrict__ src,
                                                      unsigned short* __restrict__ dst,
                                                      int n4) {
    int i = blockIdx.x * 256 + threadIdx.x;
    if (i < n4) {
        float4 x = ((const float4*)src)[i];
        ushort4 o;
        o.x = f2b(x.x); o.y = f2b(x.y); o.z = f2b(x.z); o.w = f2b(x.w);
        ((ushort4*)dst)[i] = o;
    }
}

// ---------------------------------------------------------------------------
// Per-layer static gate precompute: retention (H,dk,dk), gate (H,dk,dk)
// ---------------------------------------------------------------------------
__global__ __launch_bounds__(256) void gates_kernel(const float* __restrict__ Wl,
                                                    const float* __restrict__ V_gs,
                                                    const float* __restrict__ V_T0,
                                                    const float* __restrict__ beta_tau,
                                                    const float* __restrict__ beta_gm,
                                                    const float* __restrict__ C_ch,
                                                    const float* __restrict__ gamma_,
                                                    const float* __restrict__ alpha,
                                                    const float* __restrict__ thr,
                                                    int l,
                                                    float* __restrict__ ret,
                                                    float* __restrict__ gate) {
    int idx = blockIdx.x * 256 + threadIdx.x;     // H*dk*dk = 65536
    int h = idx >> 12;
    int lh = l * Hv + h;
    float veff = (V_gs[lh] - V_T0[lh]) + Wl[idx];
    float sp   = (veff > 20.f) ? veff : log1pf(expf(veff));
    float gch  = beta_tau[lh] * sp;
    float lam  = 1.f - expf(-gch / C_ch[lh]);
    float G    = beta_gm[lh] * sp * (1.f / (1.f + expf(-veff)));
    float sgn  = tanhf(alpha[l] * (gch - thr[l]));
    ret[idx]  = 1.f - lam;
    gate[idx] = gamma_[lh] * sgn * G;
}

// ---------------------------------------------------------------------------
// k-LayerNorm over dk=64, in place on the K section of the QKV buffer.
// ---------------------------------------------------------------------------
__global__ __launch_bounds__(256) void lnk_kernel(float* __restrict__ qkv,
                                                  const float* __restrict__ g,
                                                  const float* __restrict__ b) {
    size_t bt = blockIdx.x;
    int tid = threadIdx.x;
    int h = tid >> 4;
    int sub = tid & 15;
    float* kp = qkv + bt * (size_t)(3 * Dv) + Dv + h * DKv + sub * 4;
    float4 x = *(float4*)kp;
    float s  = x.x + x.y + x.z + x.w;
    float s2 = x.x * x.x + x.y * x.y + x.z * x.z + x.w * x.w;
    #pragma unroll
    for (int off = 1; off < 16; off <<= 1) {
        s  += __shfl_xor(s,  off, 32);
        s2 += __shfl_xor(s2, off, 32);
    }
    float mean = s * (1.f / DKv);
    float var  = s2 * (1.f / DKv) - mean * mean;
    float rstd = rsqrtf(var + EPS);
    float4 gv = ((const float4*)g)[sub];
    float4 bv = ((const float4*)b)[sub];
    float4 o;
    o.x = (x.x - mean) * rstd * gv.x + bv.x;
    o.y = (x.y - mean) * rstd * gv.y + bv.y;
    o.z = (x.z - mean) * rstd * gv.z + bv.z;
    o.w = (x.w - mean) * rstd * gv.w + bv.w;
    *(float4*)kp = o;
}

// ---------------------------------------------------------------------------
// Sequential state scan. One block per (b,h). 256 threads = 64 rows x 4
// k-groups; F/retention/gate/W rows live in registers, q/k/v staged via LDS.
// ---------------------------------------------------------------------------
__global__ __launch_bounds__(256) void scan_kernel(const float* __restrict__ qkv,
                                                   const float* __restrict__ Wl,
                                                   const float* __restrict__ ret,
                                                   const float* __restrict__ gate,
                                                   float* __restrict__ Y) {
    int bh = blockIdx.x;
    int b = bh >> 4;
    int h = bh & 15;
    int tid = threadIdx.x;
    int v  = tid & 63;
    int kg = tid >> 6;          // 0..3

    float F[16], r[16], g[16], w[16];
    int rbase = (h * DKv + v) * DKv + kg * 16;
    #pragma unroll
    for (int j = 0; j < 16; ++j) {
        F[j] = 0.f;
        r[j] = ret[rbase + j];
        g[j] = gate[rbase + j];
        w[j] = Wl[rbase + j];
    }

    __shared__ float sq[DKv], sk[DKv], sv[DKv], part[256];

    for (int t = 0; t < Tv; ++t) {
        size_t base = ((size_t)b * Tv + t) * (3 * Dv);
        if (tid < 192) {
            int sec = tid >> 6;          // 0=q 1=k 2=v
            int i = tid & 63;
            float val = qkv[base + (size_t)sec * Dv + h * DKv + i];
            if (sec == 0)      sq[i] = val;
            else if (sec == 1) sk[i] = val;
            else               sv[i] = val;
        }
        __syncthreads();
        float vv = sv[v];
        float acc = 0.f;
        #pragma unroll
        for (int j = 0; j < 16; ++j) {
            int k = kg * 16 + j;
            F[j] = r[j] * F[j] + g[j] * (vv * sk[k]);
            acc += (w[j] + F[j]) * sq[k];
        }
        part[tid] = acc;
        __syncthreads();
        if (kg == 0) {
            float y = part[v] + part[64 + v] + part[128 + v] + part[192 + v];
            Y[((size_t)b * Tv + t) * Dv + h * DKv + v] = y;
        }
    }
}

// ---------------------------------------------------------------------------
// TDM 2-D tile copy global->LDS (data_size = 2 bytes).
// Descriptor bit packing per CDNA5 ISA sec 8.3 / 8.4 (groups 2/3 zero: 2-D).
// This toolchain exposes the 6-arg builtin:
//   (uint32x4 g0, int32x8 g1, int32x4 g2, int32x4 g3, int32x8 extra, i32 cpol)
// ---------------------------------------------------------------------------
#if USE_TDM
__device__ __forceinline__ void tdm_load_2d(unsigned lds_byte_addr,
                                            const void* gaddr,
                                            unsigned tile_cols,   // dim0 (contig, elems)
                                            unsigned tile_rows,   // dim1
                                            unsigned stride_elems) {
    unsigned long long ga = (unsigned long long)gaddr;
    v4u g0;
    g0[0] = 1u;                                       // count=1, user mode
    g0[1] = lds_byte_addr;                            // LDS dest (bytes)
    g0[2] = (unsigned)(ga & 0xFFFFFFFFu);             // global_addr[31:0]
    g0[3] = (unsigned)((ga >> 32) & 0x01FFFFFFu) | (2u << 30);  // [56:32] | type=2
    v8i g1;
    g1[0] = (int)(1u << 16);                          // wg_mask=0, data_size=1 (2B)
    g1[1] = (int)((tile_cols & 0xFFFFu) << 16);       // tensor_dim0[15:0]
    g1[2] = (int)(((tile_cols >> 16) & 0xFFFFu) |
                  ((tile_rows & 0xFFFFu) << 16));     // dim0[31:16] | dim1[15:0]
    g1[3] = (int)(((tile_rows >> 16) & 0xFFFFu) |
                  ((tile_cols & 0xFFFFu) << 16));     // dim1[31:16] | tile_dim0
    g1[4] = (int)(tile_rows & 0xFFFFu);               // tile_dim1 | tile_dim2=0
    g1[5] = (int)stride_elems;                        // tensor_dim0_stride[31:0]
    g1[6] = 0;                                        // stride0[47:32] | stride1[15:0]
    g1[7] = 0;
    v4i z4 = {0, 0, 0, 0};
    v8i z8 = {0, 0, 0, 0, 0, 0, 0, 0};
    __builtin_amdgcn_tensor_load_to_lds(g0, g1, z4, z4, z8, 0);
}
#endif

// ---------------------------------------------------------------------------
// WMMA GEMM: C(MxN,f32) [+]= A(MxK bf16 rm) * W(NxK bf16 rm)^T
// Block = 256 threads = 8 waves (2x4), each wave owns a 64x64 tile
// (4x4 fragments of v_wmma_f32_16x16x32_bf16). K stepped by 32.
// TDM path: double-buffered LDS tiles (A 128x32, B 256x32); each wave DMAs
// its own 16-row A slice and 32-row B slice (tensor ops ignore EXEC, so
// per-wave slices avoid duplicate DMA issue), tracked on TENSORcnt and
// overlapped with the WMMA work of the previous K step.
// ---------------------------------------------------------------------------
__global__ __launch_bounds__(256) void gemm_bf16_wmma(const unsigned short* __restrict__ A,
                                                      const unsigned short* __restrict__ W,
                                                      float* __restrict__ C,
                                                      int N, int K, int accumulate) {
    const int tid   = threadIdx.x;
    const int wave  = tid >> 5;
    const int lane  = tid & 31;
    const int lhalf = lane >> 4;     // 0,1
    const int lmod  = lane & 15;
    const int wm = wave & 1;
    const int wn = wave >> 1;
    const int rowBase = blockIdx.x * 128 + wm * 64;
    const int colBase = blockIdx.y * 256 + wn * 64;

    v8f acc[4][4];
    if (accumulate) {
        #pragma unroll
        for (int i = 0; i < 4; ++i)
            #pragma unroll
            for (int j = 0; j < 4; ++j) {
                int rb = rowBase + i * 16 + 8 * lhalf;
                int cb = colBase + j * 16 + lmod;
                #pragma unroll
                for (int r = 0; r < 8; ++r)
                    acc[i][j][r] = C[(size_t)(rb + r) * N + cb];
            }
    } else {
        #pragma unroll
        for (int i = 0; i < 4; ++i)
            #pragma unroll
            for (int j = 0; j < 4; ++j)
                acc[i][j] = (v8f){0.f, 0.f, 0.f, 0.f, 0.f, 0.f, 0.f, 0.f};
    }

#if USE_TDM
    __shared__ unsigned short shA[2][128 * 32];   // 16 KiB
    __shared__ unsigned short shB[2][256 * 32];   // 32 KiB
    const int nk = K >> 5;
    const int aRowBlk = blockIdx.x * 128;
    const int bRowBlk = blockIdx.y * 256;

    // prologue: stage k-step 0
    {
        unsigned ldsA = (unsigned)(uintptr_t)(&shA[0][wave * 16 * 32]);
        unsigned ldsB = (unsigned)(uintptr_t)(&shB[0][wave * 32 * 32]);
        tdm_load_2d(ldsA, A + (size_t)(aRowBlk + wave * 16) * K, 32, 16, (unsigned)K);
        tdm_load_2d(ldsB, W + (size_t)(bRowBlk + wave * 32) * K, 32, 32, (unsigned)K);
    }

    for (int kb = 0; kb < nk; ++kb) {
        int buf = kb & 1;
        __builtin_amdgcn_s_wait_tensorcnt(0);   // this wave's slices landed
        __syncthreads();                        // everyone's slices visible
        if (kb + 1 < nk) {
            int k1 = (kb + 1) << 5;
            unsigned ldsA = (unsigned)(uintptr_t)(&shA[buf ^ 1][wave * 16 * 32]);
            unsigned ldsB = (unsigned)(uintptr_t)(&shB[buf ^ 1][wave * 32 * 32]);
            tdm_load_2d(ldsA, A + (size_t)(aRowBlk + wave * 16) * K + k1, 32, 16, (unsigned)K);
            tdm_load_2d(ldsB, W + (size_t)(bRowBlk + wave * 32) * K + k1, 32, 32, (unsigned)K);
        }
        FragAB a[4], bfr[4];
        #pragma unroll
        for (int i = 0; i < 4; ++i) {
            const unsigned short* ap = &shA[buf][(wm * 64 + i * 16 + lmod) * 32 + 8 * lhalf];
            a[i].q[0] = *(const uint4*)ap;
            a[i].q[1] = *(const uint4*)(ap + 16);
        }
        #pragma unroll
        for (int j = 0; j < 4; ++j) {
            const unsigned short* bp = &shB[buf][(wn * 64 + j * 16 + lmod) * 32 + 8 * lhalf];
            bfr[j].q[0] = *(const uint4*)bp;
            bfr[j].q[1] = *(const uint4*)(bp + 16);
        }
        #pragma unroll
        for (int i = 0; i < 4; ++i)
            #pragma unroll
            for (int j = 0; j < 4; ++j)
                acc[i][j] = __builtin_amdgcn_wmma_f32_16x16x32_bf16(
                    false, a[i].v, false, bfr[j].v, (short)0, acc[i][j],
                    false, false);
        __syncthreads();                        // reads done before buf is reused
    }
#else
    for (int k0 = 0; k0 < K; k0 += 32) {
        FragAB a[4], bfr[4];
        #pragma unroll
        for (int i = 0; i < 4; ++i) {
            const unsigned short* ap =
                A + (size_t)(rowBase + i * 16 + lmod) * K + k0 + 8 * lhalf;
            a[i].q[0] = *(const uint4*)ap;
            a[i].q[1] = *(const uint4*)(ap + 16);
        }
        #pragma unroll
        for (int j = 0; j < 4; ++j) {
            const unsigned short* bp =
                W + (size_t)(colBase + j * 16 + lmod) * K + k0 + 8 * lhalf;
            bfr[j].q[0] = *(const uint4*)bp;
            bfr[j].q[1] = *(const uint4*)(bp + 16);
        }
        #pragma unroll
        for (int i = 0; i < 4; ++i)
            #pragma unroll
            for (int j = 0; j < 4; ++j)
                acc[i][j] = __builtin_amdgcn_wmma_f32_16x16x32_bf16(
                    false, a[i].v, false, bfr[j].v, (short)0, acc[i][j],
                    false, false);
    }
#endif

    #pragma unroll
    for (int i = 0; i < 4; ++i)
        #pragma unroll
        for (int j = 0; j < 4; ++j) {
            int rb = rowBase + i * 16 + 8 * lhalf;
            int cb = colBase + j * 16 + lmod;
            #pragma unroll
            for (int r = 0; r < 8; ++r)
                C[(size_t)(rb + r) * N + cb] = acc[i][j][r];
        }
}

// ---------------------------------------------------------------------------
// Workspace layout (bytes)
// ---------------------------------------------------------------------------
static constexpr size_t SZ_X    = (size_t)BT * Dv * 4;        // 32 MiB
static constexpr size_t SZ_QKV  = (size_t)BT * 3 * Dv * 4;    // 96 MiB
static constexpr size_t SZ_Y    = SZ_X;                       // 32 MiB
static constexpr size_t SZ_ACTB = (size_t)BT * Dv * 2;        // 16 MiB
static constexpr size_t SZ_WB   = (size_t)Vv * Dv * 2;        // 16 MiB (max weight)
static constexpr size_t SZ_G    = (size_t)Hv * DKv * DKv * 4; // 256 KiB

static constexpr size_t OFF_X    = 0;
static constexpr size_t OFF_QKV  = OFF_X + SZ_X;
static constexpr size_t OFF_Y    = OFF_QKV + SZ_QKV;
static constexpr size_t OFF_ACTB = OFF_Y + SZ_Y;
static constexpr size_t OFF_WB   = OFF_ACTB + SZ_ACTB;
static constexpr size_t OFF_RET  = OFF_WB + SZ_WB;
static constexpr size_t OFF_GATE = OFF_RET + SZ_G;

extern "C" void kernel_launch(void* const* d_in, const int* in_sizes, int n_in,
                              void* d_out, int out_size, void* d_ws, size_t ws_size,
                              hipStream_t stream) {
    (void)in_sizes; (void)n_in; (void)out_size; (void)ws_size;
    const int*   ids      = (const int*)d_in[0];
    // d_in[1] = state_quant_bits = 16 -> quantization is identity, unused
    const float* embed_W  = (const float*)d_in[2];
    const float* conv_w   = (const float*)d_in[3];
    const float* conv_b   = (const float*)d_in[4];
    const float* Wqkv     = (const float*)d_in[5];
    const float* Wo       = (const float*)d_in[6];
    const float* W_LTM    = (const float*)d_in[7];
    const float* V_T0     = (const float*)d_in[8];
    const float* V_gs     = (const float*)d_in[9];
    const float* beta_tau = (const float*)d_in[10];
    const float* beta_gm  = (const float*)d_in[11];
    const float* C_ch     = (const float*)d_in[12];
    const float* gamma_   = (const float*)d_in[13];
    const float* alpha    = (const float*)d_in[14];
    const float* thr      = (const float*)d_in[15];
    const float* ln1_g    = (const float*)d_in[16];
    const float* ln1_b    = (const float*)d_in[17];
    const float* lnk_g    = (const float*)d_in[18];
    const float* lnk_b    = (const float*)d_in[19];
    const float* lnf_g    = (const float*)d_in[20];
    const float* lnf_b    = (const float*)d_in[21];
    const float* head_W   = (const float*)d_in[22];
    float* out = (float*)d_out;

    char* ws = (char*)d_ws;
    float*          X    = (float*)(ws + OFF_X);
    float*          QKV  = (float*)(ws + OFF_QKV);
    float*          Y    = (float*)(ws + OFF_Y);
    unsigned short* ACTB = (unsigned short*)(ws + OFF_ACTB);
    unsigned short* WB   = (unsigned short*)(ws + OFF_WB);
    float*          RET  = (float*)(ws + OFF_RET);
    float*          GATE = (float*)(ws + OFF_GATE);

    // Embedding
    embed_kernel<<<BT, 256, 0, stream>>>(ids, embed_W, X);

    for (int l = 0; l < Lv; ++l) {
        // LN1 -> Y (f32 needed for conv's temporal neighbors)
        ln1024_kernel<<<BT, 256, 0, stream>>>(X, ln1_g + l * Dv, ln1_b + l * Dv,
                                              Y, (unsigned short*)nullptr);
        // h = hln + depthwise_conv(hln), emitted as bf16 GEMM operand
        conv_kernel<<<(BT * Dv) / 256, 256, 0, stream>>>(
            Y, conv_w + (size_t)l * 3 * Dv, conv_b + (size_t)l * Dv, ACTB);

        // QKV = h @ Wqkv^T   (M=8192, N=3072, K=1024)
        cvt_f2b_kernel<<<(3 * Dv * Dv / 4 + 255) / 256, 256, 0, stream>>>(
            Wqkv + (size_t)l * 3 * Dv * Dv, WB, 3 * Dv * Dv / 4);
        gemm_bf16_wmma<<<dim3(BT / 128, (3 * Dv) / 256), 256, 0, stream>>>(
            ACTB, WB, QKV, 3 * Dv, Dv, 0);

        // k-LayerNorm in place
        lnk_kernel<<<BT, 256, 0, stream>>>(QKV, lnk_g + l * DKv, lnk_b + l * DKv);

        // Static physics gates
        gates_kernel<<<(Hv * DKv * DKv) / 256, 256, 0, stream>>>(
            W_LTM + (size_t)l * Hv * DKv * DKv, V_gs, V_T0, beta_tau, beta_gm,
            C_ch, gamma_, alpha, thr, l, RET, GATE);

        // Sequential scan over T -> Y
        scan_kernel<<<Bv * Hv, 256, 0, stream>>>(
            QKV, W_LTM + (size_t)l * Hv * DKv * DKv, RET, GATE, Y);

        // x = residual + y @ Wo^T   (accumulate into X)
        cvt_f2b_kernel<<<(BT * Dv / 4 + 255) / 256, 256, 0, stream>>>(
            Y, ACTB, BT * Dv / 4);
        cvt_f2b_kernel<<<(Dv * Dv / 4 + 255) / 256, 256, 0, stream>>>(
            Wo + (size_t)l * Dv * Dv, WB, Dv * Dv / 4);
        gemm_bf16_wmma<<<dim3(BT / 128, Dv / 256), 256, 0, stream>>>(
            ACTB, WB, X, Dv, Dv, 1);
    }

    // Final LN -> bf16 operand
    ln1024_kernel<<<BT, 256, 0, stream>>>(X, lnf_g, lnf_b, Y, ACTB);

    // logits = xf @ head_W^T   (M=8192, N=8192, K=1024)
    cvt_f2b_kernel<<<(Vv * Dv / 4 + 255) / 256, 256, 0, stream>>>(
        head_W, WB, Vv * Dv / 4);
    gemm_bf16_wmma<<<dim3(BT / 128, Vv / 256), 256, 0, stream>>>(
        ACTB, WB, out, Vv, Dv, 0);
}
